// MLPSpeculatorModel_63745904607334
// MI455X (gfx1250) — compile-verified, matrix operand order checked by reading, not run on hardware
//
#include <hip/hip_runtime.h>
#include <hip/hip_bf16.h>
#include <math.h>

typedef __attribute__((ext_vector_type(16))) __bf16 v16bf;
typedef __attribute__((ext_vector_type(8)))  __bf16 v8bf;
typedef __attribute__((ext_vector_type(8)))  float  v8f;
typedef __attribute__((ext_vector_type(4)))  float  v4f;

// LDS A-tile: 256 rows x 32 K bf16, padded row stride of 40 elements (80 B,
// 16B-aligned for ds_load_b128, 20-dword stride to spread banks).
#define LDS_ROW   40
#define LDS_STAGE (256 * LDS_ROW)   // elements per stage

// ---------------------------------------------------------------------------
// Async stage of one 256x32 bf16 A-chunk into LDS (GLOBAL_LOAD_ASYNC_TO_LDS,
// ASYNCcnt-tracked). Thread t copies row t: 4 x 16B chunks.
// ---------------------------------------------------------------------------
__device__ inline void stage_A_async(const __bf16* __restrict__ A, int lda,
                                     int k0, __bf16* dst_lds) {
    int tid = (int)threadIdx.x;                       // row index 0..255
    const __bf16* g = A + (size_t)tid * (size_t)lda + (size_t)k0;
    unsigned int l = (unsigned int)(unsigned long long)dst_lds
                   + (unsigned int)tid * (LDS_ROW * 2u);
#pragma unroll
    for (int c = 0; c < 4; ++c) {
        asm volatile("global_load_async_to_lds_b128 %0, %1, off"
                     :: "v"(l + (unsigned int)c * 16u), "v"(g + c * 8)
                     : "memory");
    }
}

__device__ inline void wait_async_all() {
    asm volatile("s_wait_asynccnt 0" ::: "memory");
}

// ---------------------------------------------------------------------------
// 16x32 bf16 A-fragment from the padded LDS tile (ISA 7.12.2 layout):
// lane L holds row (L&15); lanes 0-15: K=[0..7],[16..23]; lanes 16-31:
// K=[8..15],[24..31] -> two 16B ds_load_b128 per lane.
// ---------------------------------------------------------------------------
__device__ inline v16bf lds_frag(const __bf16* sbase, int r0) {
    int lane = (int)threadIdx.x & 31;
    int r    = lane & 15;
    int kh   = (lane >> 4) << 3;      // 0 or 8
    const __bf16* p = sbase + (r0 + r) * LDS_ROW + kh;
    v8bf a = *(const v8bf*)(p);        // K = kh    .. kh+7
    v8bf b = *(const v8bf*)(p + 16);   // K = kh+16 .. kh+23
    v16bf out;
#pragma unroll
    for (int t = 0; t < 8; ++t) { out[t] = a[t]; out[8 + t] = b[t]; }
    return out;
}

// ---------------------------------------------------------------------------
// Raw f32 weight fragment (non-temporal HBM stream, kept in 16 VGPRs so the
// k+1 fragment can be prefetched while k's 16 WMMAs execute), plus the
// bf16 conversion done at use time (VALU is free; kernel is HBM-bound).
// ---------------------------------------------------------------------------
struct RawB { v4f a0, a1, b0, b1; };

__device__ inline RawB load_rawB(const float* __restrict__ src,
                                 int ld, int r0, int k0) {
    int lane = (int)threadIdx.x & 31;
    int r    = lane & 15;
    int kh   = (lane >> 4) << 3;
    const float* p = src + (size_t)(r0 + r) * (size_t)ld + (size_t)(k0 + kh);
    RawB rb;
    rb.a0 = __builtin_nontemporal_load((const v4f*)(p));
    rb.a1 = __builtin_nontemporal_load((const v4f*)(p + 4));
    rb.b0 = __builtin_nontemporal_load((const v4f*)(p + 16));
    rb.b1 = __builtin_nontemporal_load((const v4f*)(p + 20));
    return rb;
}

__device__ inline v16bf cvtB(const RawB& rb) {
    v16bf out;
#pragma unroll
    for (int t = 0; t < 4; ++t) {
        out[t]      = (__bf16)rb.a0[t];
        out[4 + t]  = (__bf16)rb.a1[t];
        out[8 + t]  = (__bf16)rb.b0[t];
        out[12 + t] = (__bf16)rb.b1[t];
    }
    return out;
}

// ---------------------------------------------------------------------------
// GEMM: out[m][n] = sum_k A[m][k] * Bw[n][k]      (M fixed = 256)
// Block = 8 waves; wave owns 16 N-columns and ALL 16 M-tiles (K outer), so
// each streamed weight fragment is reused 16x. Three-level pipelining:
//   - A chunks double-buffered in LDS via async loads (stage k+1 during k)
//   - raw f32 weight fragment for k+1 prefetched into registers during k
//   - A-fragment ds_loads for tile m+1 issued before WMMA of tile m
// ---------------------------------------------------------------------------
__global__ __launch_bounds__(256)
void gemm_bf16_kernel(const __bf16* __restrict__ A, int lda,
                      const float* __restrict__ Bw,
                      float* __restrict__ out, int N, int K) {
    __shared__ __bf16 smem[2 * LDS_STAGE];    // 2 x 20 KB stages

    int wave = (int)(threadIdx.x >> 5);
    int n0   = (int)blockIdx.x * 128 + wave * 16;

    v8f acc[16];
#pragma unroll
    for (int m = 0; m < 16; ++m) acc[m] = (v8f)(0.0f);

    // Prologue: stage A chunk 0, prefetch weight fragment 0.
    RawB braw = load_rawB(Bw, K, n0, 0);
    stage_A_async(A, lda, 0, smem);
    wait_async_all();
    __syncthreads();

    int nk  = K >> 5;
    int cur = 0;
    for (int ks = 0; ks < nk; ++ks) {
        if (ks + 1 < nk)
            stage_A_async(A, lda, (ks + 1) << 5, smem + (cur ^ 1) * LDS_STAGE);

        v16bf bfrag = cvtB(braw);                     // waits on k's loads
        if (ks + 1 < nk)
            braw = load_rawB(Bw, K, n0, (ks + 1) << 5);  // k+1 HBM prefetch

        const __bf16* sbase = smem + cur * LDS_STAGE;
        v16bf a_cur = lds_frag(sbase, 0);
#pragma unroll
        for (int m = 0; m < 16; ++m) {
            v16bf a_next;
            if (m < 15) a_next = lds_frag(sbase, (m + 1) * 16);
            acc[m] = __builtin_amdgcn_wmma_f32_16x16x32_bf16(
                false, a_cur, false, bfrag, (short)0, acc[m], false, false);
            if (m < 15) a_cur = a_next;
        }

        wait_async_all();     // my async writes to the other stage are done
        __syncthreads();      // everyone done reading cur + next stage visible
        cur ^= 1;
    }

    // C/D layout: VGPR r -> row (r + (lane>=16 ? 8:0)), col = lane&15
    int lane    = (int)threadIdx.x & 31;
    int col     = n0 + (lane & 15);
    int rowbase = (lane >> 4) * 8;
#pragma unroll
    for (int m = 0; m < 16; ++m) {
#pragma unroll
        for (int r = 0; r < 8; ++r) {
            int row = m * 16 + rowbase + r;
            out[(size_t)row * (size_t)N + (size_t)col] = acc[m][r];
        }
    }
}

// ---------------------------------------------------------------------------
// Per-row: s = sw*pre + ec*emb[ind[b]]; RMSNorm; gelu(ln_w*xn+ln_b) exact;
// write bf16 state for the next WMMA GEMMs. One block per batch row.
// ---------------------------------------------------------------------------
__global__ __launch_bounds__(256)
void fuse_rmsnorm_gelu_kernel(const float* __restrict__ pre,
                              const float* __restrict__ emb,   // [VOCAB][2048]
                              const int* __restrict__ ind,
                              const float* __restrict__ lnw,
                              const float* __restrict__ lnb,
                              __bf16* __restrict__ state_out,  // [256][2048]
                              float sw, float ec) {
    __shared__ float red[256];
    int b   = (int)blockIdx.x;
    int tid = (int)threadIdx.x;
    const float* prow = pre + (size_t)b * 2048;
    const float* erow = emb + (size_t)ind[b] * 2048;

    float s[8];
    float ss = 0.0f;
#pragma unroll
    for (int t = 0; t < 8; ++t) {
        int j   = tid + t * 256;
        float v = sw * prow[j] + ec * erow[j];
        s[t] = v;
        ss  += v * v;
    }
    red[tid] = ss;
    __syncthreads();
    for (int off = 128; off > 0; off >>= 1) {
        if (tid < off) red[tid] += red[tid + off];
        __syncthreads();
    }
    float rinv = rsqrtf(red[0] * (1.0f / 2048.0f) + 1e-6f);

    __bf16* orow = state_out + (size_t)b * 2048;
#pragma unroll
    for (int t = 0; t < 8; ++t) {
        int j    = tid + t * 256;
        float xn = s[t] * rinv;
        float x  = lnw[j] * xn + lnb[j];
        float g  = 0.5f * x * (1.0f + erff(x * 0.70710678118654752f));
        orow[j]  = (__bf16)g;
    }
}

// ---------------------------------------------------------------------------
// Softmax over 32000 + argmax (first-occurrence tie-break, matching jnp).
// probs -> d_out[b][step][v]; argmax -> ind[b]. One block per batch row.
// ---------------------------------------------------------------------------
__global__ __launch_bounds__(256)
void softmax_argmax_kernel(const float* __restrict__ logits,  // [256][32000]
                           float* __restrict__ out,           // [256][3][32000]
                           int* __restrict__ ind, int step) {
    __shared__ float vmax[256];
    __shared__ int   vidx[256];
    __shared__ float vsum[256];
    int b   = (int)blockIdx.x;
    int tid = (int)threadIdx.x;
    const float* row = logits + (size_t)b * 32000;

    float m = -3.4e38f; int mi = 0x7fffffff;
    for (int j = tid; j < 32000; j += 256) {
        float v = row[j];
        if (v > m) { m = v; mi = j; }
    }
    vmax[tid] = m; vidx[tid] = mi;
    __syncthreads();
    for (int off = 128; off > 0; off >>= 1) {
        if (tid < off) {
            float vo = vmax[tid + off]; int io = vidx[tid + off];
            if (vo > vmax[tid] || (vo == vmax[tid] && io < vidx[tid])) {
                vmax[tid] = vo; vidx[tid] = io;
            }
        }
        __syncthreads();
    }
    float rmax = vmax[0];

    float s = 0.0f;
    for (int j = tid; j < 32000; j += 256) s += expf(row[j] - rmax);
    vsum[tid] = s;
    __syncthreads();
    for (int off = 128; off > 0; off >>= 1) {
        if (tid < off) vsum[tid] += vsum[tid + off];
        __syncthreads();
    }
    float rinv = 1.0f / vsum[0];

    float* orow = out + ((size_t)b * 3 + (size_t)step) * 32000;
    for (int j = tid; j < 32000; j += 256) orow[j] = expf(row[j] - rmax) * rinv;
    if (tid == 0) ind[b] = vidx[0];
}

// ---------------------------------------------------------------------------
// Init: f32 hidden -> bf16 A-matrix; int64 ids -> int32 ind.
// ---------------------------------------------------------------------------
__global__ __launch_bounds__(256)
void cvt_hidden_kernel(const float* __restrict__ h, __bf16* __restrict__ o, int n) {
    int idx = (int)blockIdx.x * 256 + (int)threadIdx.x;
    if (idx < n) o[idx] = (__bf16)h[idx];
}

__global__ void cvt_ids_kernel(const long long* __restrict__ ids,
                               int* __restrict__ ind) {
    ind[threadIdx.x] = (int)ids[threadIdx.x];
}

// ---------------------------------------------------------------------------
extern "C" void kernel_launch(void* const* d_in, const int* in_sizes, int n_in,
                              void* d_out, int out_size, void* d_ws, size_t ws_size,
                              hipStream_t stream) {
    const float*     hidden = (const float*)d_in[0];
    const long long* ids    = (const long long*)d_in[1];
    const float*     emb_w  = (const float*)d_in[2];
    const float*     proj0  = (const float*)d_in[3];
    const float*     projr  = (const float*)d_in[4];
    const float*     head_w = (const float*)d_in[5];
    const float*     ln_w   = (const float*)d_in[6];
    const float*     ln_b   = (const float*)d_in[7];
    float* out = (float*)d_out;

    const int B = 256, H = 4096, I = 2048, V = 32000, NP = 3;

    // Workspace layout (all 256B-aligned):
    char*   ws       = (char*)d_ws;
    int*    ind      = (int*)ws;                                   //   1 KB
    __bf16* state_bf = (__bf16*)(ws + 4096);                       //   2 MB
    float*  pre      = (float*)(ws + 4096 + (size_t)B * H * 2);    //   2 MB
    float*  logits   = (float*)(ws + 4096 + (size_t)B * H * 2
                                     + (size_t)B * I * 4);         // 32.8 MB

    double sw  = pow(0.5, 0.5 / (double)NP);
    float  swf = (float)sw;
    float  ecf = (float)(sqrt(1.0 - sw * sw) * sqrt((double)I / 2.0));

    cvt_ids_kernel<<<1, 256, 0, stream>>>(ids, ind);
    cvt_hidden_kernel<<<(B * H + 255) / 256, 256, 0, stream>>>(hidden, state_bf, B * H);

    for (int i = 0; i < NP; ++i) {
        const float* pw = (i == 0) ? proj0 : (projr + (size_t)(i - 1) * I * I);
        int K = (i == 0) ? H : I;

        // pre = state @ pw^T   (M=256, N=2048)
        gemm_bf16_kernel<<<I / 128, 256, 0, stream>>>(state_bf, K, pw, pre, I, K);

        // s = sw*pre + ec*emb[ind]; RMSNorm; gelu -> bf16 state (ld = 2048)
        fuse_rmsnorm_gelu_kernel<<<B, 256, 0, stream>>>(
            pre, emb_w + (size_t)i * V * I, ind,
            ln_w + (size_t)i * I, ln_b + (size_t)i * I, state_bf, swf, ecf);

        // logits = state @ head_w[i]^T   (M=256, N=32000) — HBM-bound stream
        gemm_bf16_kernel<<<V / 128, 256, 0, stream>>>(
            state_bf, I, head_w + (size_t)i * V * I, logits, V, I);

        // probs -> d_out[:, i, :], argmax -> ind
        softmax_argmax_kernel<<<B, 256, 0, stream>>>(logits, out, ind, i);
    }
}
